// DASAttentionGate_20340965114394
// MI455X (gfx1250) — matrix operand order repaired and checked by reading, not compiled, not run on hardware
//
#include <hip/hip_runtime.h>
#include <hip/hip_bf16.h>

// ---------- CDNA5 WMMA types ----------
typedef __attribute__((ext_vector_type(16))) __bf16 v16bf;
typedef __attribute__((ext_vector_type(8)))  float  v8f;

union ABFrag {
    v16bf v;
    unsigned short u[16];
    uint4 q[2];
};

// Optional CDNA5 async global->LDS path (ASYNCcnt), with safe fallback.
#if __has_builtin(__builtin_amdgcn_global_load_async_to_lds_b128) && \
    __has_builtin(__builtin_amdgcn_s_wait_asynccnt)
#define HAVE_ASYNC_LDS 1
#else
#define HAVE_ASYNC_LDS 0
#endif

#if HAVE_ASYNC_LDS
// From the compiler diagnostic: param 0 is a pointer to a 16-byte int vector
// in the global (device) address space; param 1 is the LDS-side counterpart.
typedef int vi4_t __attribute__((vector_size(16)));
#endif

__device__ __forceinline__ void copy16_g2l(const unsigned short* g, unsigned short* l) {
#if HAVE_ASYNC_LDS
    __builtin_amdgcn_global_load_async_to_lds_b128(
        (__attribute__((address_space(1))) vi4_t*)(g),
        (__attribute__((address_space(3))) vi4_t*)(l),
        0, 0);
#else
    *reinterpret_cast<uint4*>(l) = *reinterpret_cast<const uint4*>(g);
#endif
}
__device__ __forceinline__ void async_copy_fence() {
#if HAVE_ASYNC_LDS
    __builtin_amdgcn_s_wait_asynccnt(0);
#endif
}

// ---------- problem constants ----------
#define BATCH 4
#define CIN   128
#define COUT  128
#define HH    96
#define WW    96
#define PP    (HH * WW)          // 9216 pixels per image
#define KTAP  9
#define KDEF  (CIN * KTAP)       // 1152 = deform/offset GEMM K
#define NT_PW 64                 // pointwise N tile (pixels per block)
#define EPSN  1e-5f

// ---------- bf16 helpers (bit-level, RNE) ----------
__device__ __forceinline__ unsigned short f32_to_bf16u(float f) {
    unsigned int u = __float_as_uint(f);
    u += 0x7FFFu + ((u >> 16) & 1u);
    return (unsigned short)(u >> 16);
}
__device__ __forceinline__ float bf16u_to_f32(unsigned short h) {
    return __uint_as_float(((unsigned int)h) << 16);
}

// =====================================================================
// 1) prep: bf16-pack pw_w [o][c]; de_w -> [o][k*128+c]; off_w -> padded
//    [32][k*128+c] (rows 18..31 zero); zero GroupNorm accumulators
// =====================================================================
__global__ void k_prep(const float* __restrict__ pw_w,
                       const float* __restrict__ de_w,
                       const float* __restrict__ off_w,
                       unsigned short* __restrict__ pwA,
                       unsigned short* __restrict__ wA,
                       unsigned short* __restrict__ offA,
                       float* __restrict__ stats) {
    int idx = blockIdx.x * blockDim.x + threadIdx.x;
    if (idx < COUT * KDEF) {
        int o = idx / KDEF;
        int r = idx - o * KDEF;
        int k = r >> 7;           // r / 128
        int c = r & 127;          // r % 128
        wA[idx] = f32_to_bf16u(de_w[((size_t)o * CIN + c) * KTAP + k]);
    }
    if (idx < 32 * KDEF) {        // 36864, M padded 18 -> 32
        int o = idx / KDEF;
        int r = idx - o * KDEF;
        int k = r >> 7;
        int c = r & 127;
        offA[idx] = (o < 18) ? f32_to_bf16u(off_w[((size_t)o * CIN + c) * KTAP + k])
                             : (unsigned short)0;
    }
    if (idx < COUT * CIN) {
        pwA[idx] = f32_to_bf16u(pw_w[idx]);   // [o][c] row-major already
    }
    if (idx < 8) stats[idx] = 0.0f;
}

// =====================================================================
// 2) depthwise 3x3 + bias -> bf16 h0 [b][c][p]
// =====================================================================
__global__ void k_dw(const float* __restrict__ x,
                     const float* __restrict__ dw_w,
                     const float* __restrict__ dw_b,
                     unsigned short* __restrict__ h0) {
    size_t idx = (size_t)blockIdx.x * blockDim.x + threadIdx.x;
    if (idx >= (size_t)BATCH * CIN * PP) return;
    int p = (int)(idx % PP);
    int bc = (int)(idx / PP);
    int c = bc & (CIN - 1);
    int y = p / WW, xx = p - y * WW;
    const float* xin = x + (size_t)bc * PP;
    const float* wv = dw_w + c * 9;
    float acc = dw_b[c];
#pragma unroll
    for (int ky = -1; ky <= 1; ++ky) {
        int yy = y + ky;
        if (yy < 0 || yy >= HH) continue;
#pragma unroll
        for (int kx = -1; kx <= 1; ++kx) {
            int xw = xx + kx;
            if (xw < 0 || xw >= WW) continue;
            acc += xin[yy * WW + xw] * wv[(ky + 1) * 3 + (kx + 1)];
        }
    }
    h0[idx] = f32_to_bf16u(acc);
}

// =====================================================================
// 3) pointwise GEMM via WMMA, LDS-staged B panel (async where available):
//    sc[b][o][p] = sum_c pwA[o][c]*h0[b][c][p] + pw_b[o]
//    grid = (P/64, B), block = 256 (8 waves); wave w -> rows 16w..16w+15,
//    4 N-subtiles of 16 px each (A fragment reused across subtiles).
// =====================================================================
__global__ void k_pw_wmma(const unsigned short* __restrict__ h0,
                          const unsigned short* __restrict__ pwA,
                          const float* __restrict__ pw_b,
                          float* __restrict__ sc) {
    __shared__ __align__(16) unsigned short Bt[CIN * NT_PW];   // 16 KB
    const int b  = blockIdx.y;
    const int p0 = blockIdx.x * NT_PW;
    const int t  = threadIdx.x;
    const unsigned short* Bb = h0 + (size_t)b * CIN * PP;

    // stage 128 rows x 64 px bf16 (16 KB) = 1024 x 16B chunks, 4 per thread
#pragma unroll
    for (int i = 0; i < 4; ++i) {
        int j = t + i * 256;
        int row = j >> 3;          // 0..127 (channel)
        int seg = j & 7;           // 8 x 16B per row
        copy16_g2l(Bb + (size_t)row * PP + p0 + seg * 8,
                   Bt + row * NT_PW + seg * 8);
    }
    async_copy_fence();
    __syncthreads();

    const int lane = t & 31;
    const int wv   = t >> 5;
    const int o0   = wv * 16;
    const int m     = lane & 15;
    const int base0 = (lane < 16) ? 0 : 8;
    const int base1 = base0 + 16;

    v8f acc0 = {0.f,0.f,0.f,0.f,0.f,0.f,0.f,0.f};
    v8f acc1 = acc0, acc2 = acc0, acc3 = acc0;
#pragma unroll
    for (int kk = 0; kk < CIN; kk += 32) {
        ABFrag a;
        const unsigned short* arow = pwA + (size_t)(o0 + m) * CIN + kk;
        a.q[0] = *reinterpret_cast<const uint4*>(arow + base0);
        a.q[1] = *reinterpret_cast<const uint4*>(arow + base1);
        ABFrag b0, b1, b2, b3;
#pragma unroll
        for (int s = 0; s < 16; ++s) {
            const int kr = kk + ((s < 8) ? (base0 + s) : (base1 + s - 8));
            const unsigned short* brow = Bt + kr * NT_PW + (lane & 15);  // B col N=lane%16
            b0.u[s] = brow[0];
            b1.u[s] = brow[16];
            b2.u[s] = brow[32];
            b3.u[s] = brow[48];
        }
        acc0 = __builtin_amdgcn_wmma_f32_16x16x32_bf16(false, a.v, false, b0.v, (short)0, acc0, false, false);
        acc1 = __builtin_amdgcn_wmma_f32_16x16x32_bf16(false, a.v, false, b1.v, (short)0, acc1, false, false);
        acc2 = __builtin_amdgcn_wmma_f32_16x16x32_bf16(false, a.v, false, b2.v, (short)0, acc2, false, false);
        acc3 = __builtin_amdgcn_wmma_f32_16x16x32_bf16(false, a.v, false, b3.v, (short)0, acc3, false, false);
    }
    const int n  = lane & 15;
    const int mo = (lane < 16) ? 0 : 8;
#pragma unroll
    for (int r = 0; r < 8; ++r) {
        int o = o0 + mo + r;
        float bias = pw_b[o];
        float* orow = sc + ((size_t)b * COUT + o) * PP + p0 + n;
        orow[0]  = acc0[r] + bias;
        orow[16] = acc1[r] + bias;
        orow[32] = acc2[r] + bias;
        orow[48] = acc3[r] + bias;
    }
}

// =====================================================================
// 4) InstanceNorm (biased var) + ReLU in-place on sc; also channel-last bf16
//    grid = (C, B), block = 256
// =====================================================================
__global__ void k_inorm(float* __restrict__ sc,
                        unsigned short* __restrict__ h_cl) {
    __shared__ float s_sum[256];
    __shared__ float s_sq[256];
    const int b = blockIdx.y, c = blockIdx.x, t = threadIdx.x;
    float* row = sc + ((size_t)b * COUT + c) * PP;
    float s = 0.f, ss = 0.f;
    for (int p = t; p < PP; p += 256) {
        float v = row[p];
        s += v; ss += v * v;
    }
    s_sum[t] = s; s_sq[t] = ss;
    __syncthreads();
    for (int off = 128; off > 0; off >>= 1) {
        if (t < off) { s_sum[t] += s_sum[t + off]; s_sq[t] += s_sq[t + off]; }
        __syncthreads();
    }
    const float mu = s_sum[0] * (1.0f / PP);
    const float var = s_sq[0] * (1.0f / PP) - mu * mu;
    const float rstd = rsqrtf(var + EPSN);
    for (int p = t; p < PP; p += 256) {
        float v = (row[p] - mu) * rstd;
        v = fmaxf(v, 0.f);
        row[p] = v;                                              // shortcut (f32)
        h_cl[((size_t)b * PP + p) * COUT + c] = f32_to_bf16u(v); // [b][y][x][c]
    }
}

// =====================================================================
// 5) offset conv 128->18 as WMMA GEMM (M padded to 32):
//    B[k*128+c][pix] = h(y+ky, x+kx, c) from channel-last bf16 (zero pad).
//    grid = (P/16, B), block = 256; waves 0,1 run the GEMM.
// =====================================================================
__global__ void k_offc_wmma(const unsigned short* __restrict__ h_cl,
                            const unsigned short* __restrict__ offA,
                            const float* __restrict__ off_b,
                            float* __restrict__ offbuf) {
    __shared__ __align__(16) unsigned short Bt[KDEF * 16];   // 36 KB
    const int b  = blockIdx.y;
    const int p0 = blockIdx.x * 16;
    const int t  = threadIdx.x;

    if (t < 144) {
        const int pix = t / 9;
        const int k   = t - pix * 9;
        const int p = p0 + pix;
        const int y = p / WW, x = p - y * WW;
        const int yy = y + (k / 3 - 1);
        const int xx = x + (k % 3 - 1);
        const bool valid = (yy >= 0) & (yy < HH) & (xx >= 0) & (xx < WW);
        const unsigned short* src =
            h_cl + ((size_t)b * PP + (size_t)(valid ? yy : 0) * WW + (valid ? xx : 0)) * COUT;
        for (int c = 0; c < CIN; ++c) {
            Bt[(k * CIN + c) * 16 + pix] = valid ? src[c] : (unsigned short)0;
        }
    }
    __syncthreads();

    const int lane = t & 31;
    const int wv   = t >> 5;
    if (wv < 2) {                    // wave-uniform branch; EXEC stays all-1s
        const int o0 = wv * 16;
        const int m     = lane & 15;
        const int base0 = (lane < 16) ? 0 : 8;
        const int base1 = base0 + 16;
        v8f acc = {0.f,0.f,0.f,0.f,0.f,0.f,0.f,0.f};
        for (int kt = 0; kt < KDEF; kt += 32) {
            ABFrag a, bb;
            const unsigned short* arow = offA + (size_t)(o0 + m) * KDEF + kt;
            a.q[0] = *reinterpret_cast<const uint4*>(arow + base0);
            a.q[1] = *reinterpret_cast<const uint4*>(arow + base1);
#pragma unroll
            for (int s = 0; s < 16; ++s) {
                const int kr = kt + ((s < 8) ? (base0 + s) : (base1 + s - 8));
                bb.u[s] = Bt[kr * 16 + (lane & 15)];
            }
            acc = __builtin_amdgcn_wmma_f32_16x16x32_bf16(
                false, a.v, false, bb.v, (short)0, acc, false, false);
        }
        const int n  = lane & 15;
        const int mo = (lane < 16) ? 0 : 8;
#pragma unroll
        for (int r = 0; r < 8; ++r) {
            int oc = o0 + mo + r;
            if (oc < 18)
                offbuf[((size_t)b * 18 + oc) * PP + p0 + n] = acc[r] + off_b[oc];
        }
    }
}

// =====================================================================
// 6) deformable conv: stage bilinear samples (16 pixels x 1152) into LDS,
//    then K=1152 WMMA GEMM. grid = (P/16, B), block = 256 (8 waves).
// =====================================================================
__global__ void k_deform_wmma(const unsigned short* __restrict__ h_cl,
                              const float* __restrict__ offbuf,
                              const unsigned short* __restrict__ wA,
                              const float* __restrict__ de_b,
                              float* __restrict__ dbuf) {
    __shared__ __align__(16) unsigned short samp[KDEF * 16];   // 36 KB
    const int b  = blockIdx.y;
    const int p0 = blockIdx.x * 16;
    const int t  = threadIdx.x;

    if (t < 144) {
        const int pix = t / 9;
        const int k   = t - pix * 9;
        const int p = p0 + pix;
        const int y = p / WW, x = p - y * WW;
        const float dy = offbuf[((size_t)b * 18 + 2 * k) * PP + p];
        const float dx = offbuf[((size_t)b * 18 + 2 * k + 1) * PP + p];
        const float py = (float)y + (float)(k / 3 - 1) + dy;
        const float px = (float)x + (float)(k % 3 - 1) + dx;
        const float y0f = floorf(py), x0f = floorf(px);
        const float ly = py - y0f, lx = px - x0f;
        const int y0 = (int)y0f, x0 = (int)x0f;
        const int y1 = y0 + 1,  x1 = x0 + 1;
        const bool vy0 = (y0 >= 0) & (y0 < HH), vy1 = (y1 >= 0) & (y1 < HH);
        const bool vx0 = (x0 >= 0) & (x0 < WW), vx1 = (x1 >= 0) & (x1 < WW);
        float w00 = (1.f - ly) * (1.f - lx) * ((vy0 & vx0) ? 1.f : 0.f);
        float w01 = (1.f - ly) * lx         * ((vy0 & vx1) ? 1.f : 0.f);
        float w10 = ly * (1.f - lx)         * ((vy1 & vx0) ? 1.f : 0.f);
        float w11 = ly * lx                 * ((vy1 & vx1) ? 1.f : 0.f);
        const int y0c = min(max(y0, 0), HH - 1), y1c = min(max(y1, 0), HH - 1);
        const int x0c = min(max(x0, 0), WW - 1), x1c = min(max(x1, 0), WW - 1);
        const unsigned short* hb = h_cl + (size_t)b * PP * COUT;
        const unsigned short* p00 = hb + ((size_t)y0c * WW + x0c) * COUT;
        const unsigned short* p01 = hb + ((size_t)y0c * WW + x1c) * COUT;
        const unsigned short* p10 = hb + ((size_t)y1c * WW + x0c) * COUT;
        const unsigned short* p11 = hb + ((size_t)y1c * WW + x1c) * COUT;
        for (int c = 0; c < CIN; ++c) {
            float v = w00 * bf16u_to_f32(p00[c]) + w01 * bf16u_to_f32(p01[c])
                    + w10 * bf16u_to_f32(p10[c]) + w11 * bf16u_to_f32(p11[c]);
            samp[(k * CIN + c) * 16 + pix] = f32_to_bf16u(v);
        }
    }
    __syncthreads();

    const int lane = t & 31;
    const int wv   = t >> 5;
    const int o0   = wv * 16;
    const int m     = lane & 15;
    const int base0 = (lane < 16) ? 0 : 8;
    const int base1 = base0 + 16;

    v8f acc = {0.f,0.f,0.f,0.f,0.f,0.f,0.f,0.f};
    for (int kt = 0; kt < KDEF; kt += 32) {
        ABFrag a, bb;
        const unsigned short* arow = wA + (size_t)(o0 + m) * KDEF + kt;
        a.q[0] = *reinterpret_cast<const uint4*>(arow + base0);
        a.q[1] = *reinterpret_cast<const uint4*>(arow + base1);
#pragma unroll
        for (int s = 0; s < 16; ++s) {
            const int kr = kt + ((s < 8) ? (base0 + s) : (base1 + s - 8));
            bb.u[s] = samp[kr * 16 + (lane & 15)];
        }
        acc = __builtin_amdgcn_wmma_f32_16x16x32_bf16(
            false, a.v, false, bb.v, (short)0, acc, false, false);
    }
    const int n  = lane & 15;
    const int mo = (lane < 16) ? 0 : 8;
#pragma unroll
    for (int r = 0; r < 8; ++r) {
        int o = o0 + mo + r;
        dbuf[((size_t)b * COUT + o) * PP + p0 + n] = acc[r] + de_b[o];
    }
}

// =====================================================================
// 7) GroupNorm(1,C) partial reduce: per-batch sum / sumsq via LDS + atomics
// =====================================================================
__global__ void k_gstat(const float* __restrict__ dbuf,
                        float* __restrict__ stats) {
    __shared__ float s_sum[256];
    __shared__ float s_sq[256];
    const int b = blockIdx.y, t = threadIdx.x;
    size_t base = (size_t)b * COUT * PP + (size_t)blockIdx.x * 4096 + t;
    float s = 0.f, ss = 0.f;
#pragma unroll
    for (int i = 0; i < 16; ++i) {
        float v = dbuf[base + (size_t)i * 256];
        s += v; ss += v * v;
    }
    s_sum[t] = s; s_sq[t] = ss;
    __syncthreads();
    for (int off = 128; off > 0; off >>= 1) {
        if (t < off) { s_sum[t] += s_sum[t + off]; s_sq[t] += s_sq[t + off]; }
        __syncthreads();
    }
    if (t == 0) {
        atomicAdd(&stats[b * 2 + 0], s_sum[0]);
        atomicAdd(&stats[b * 2 + 1], s_sq[0]);
    }
}

// 8) finalize per-batch mean / rstd
__global__ void k_gfin(const float* __restrict__ stats, float* __restrict__ mr) {
    int t = threadIdx.x;
    if (t < BATCH) {
        const float inv_n = 1.0f / ((float)COUT * (float)PP);
        float mu = stats[t * 2 + 0] * inv_n;
        float var = stats[t * 2 + 1] * inv_n - mu * mu;
        mr[t * 2 + 0] = mu;
        mr[t * 2 + 1] = rsqrtf(var + EPSN);
    }
}

// 9) gate: out = shortcut * (1 + sigmoid(gn(d)))
__global__ void k_gate(const float* __restrict__ dbuf,
                       const float* __restrict__ sc,
                       const float* __restrict__ gn_w,
                       const float* __restrict__ gn_b,
                       const float* __restrict__ mr,
                       float* __restrict__ out) {
    size_t idx = (size_t)blockIdx.x * blockDim.x + threadIdx.x;
    if (idx >= (size_t)BATCH * COUT * PP) return;
    int bc = (int)(idx / PP);
    int c = bc & (COUT - 1);
    int b = bc >> 7;
    float mu = mr[b * 2 + 0], rstd = mr[b * 2 + 1];
    float g = (dbuf[idx] - mu) * rstd * gn_w[c] + gn_b[c];
    float sig = 1.0f / (1.0f + __expf(-g));
    out[idx] = sc[idx] * (1.0f + sig);
}

// =====================================================================
// launch
// =====================================================================
extern "C" void kernel_launch(void* const* d_in, const int* in_sizes, int n_in,
                              void* d_out, int out_size, void* d_ws, size_t ws_size,
                              hipStream_t stream) {
    const float* x_in  = (const float*)d_in[0];
    const float* dw_w  = (const float*)d_in[1];
    const float* dw_b  = (const float*)d_in[2];
    const float* pw_w  = (const float*)d_in[3];
    const float* pw_b  = (const float*)d_in[4];
    const float* off_w = (const float*)d_in[5];
    const float* off_b = (const float*)d_in[6];
    const float* de_w  = (const float*)d_in[7];
    const float* de_b  = (const float*)d_in[8];
    const float* gn_w  = (const float*)d_in[9];
    const float* gn_b  = (const float*)d_in[10];
    float* out = (float*)d_out;

    // workspace carve (offsets 256B-aligned except mr, which is 32B-aligned)
    char* ws = (char*)d_ws;
    unsigned short* h0   = (unsigned short*)(ws + 0);              //  9,437,184 B
    unsigned short* pwA  = (unsigned short*)(ws + 9437184);        //     32,768 B
    float*          sc   = (float*)(ws + 9469952);                 // 18,874,368 B
    unsigned short* h_cl = (unsigned short*)(ws + 28344320);       //  9,437,184 B
    float*          offb = (float*)(ws + 37781504);                //  2,654,208 B
    unsigned short* wA   = (unsigned short*)(ws + 40435712);       //    294,912 B
    unsigned short* offA = (unsigned short*)(ws + 40730624);       //     73,728 B
    float*          dbuf = (float*)(ws + 40804352);                // 18,874,368 B
    float*          stats= (float*)(ws + 59678720);                //  32 B
    float*          mr   = (float*)(ws + 59678752);                //  32 B

    const size_t nElem = (size_t)BATCH * COUT * PP;   // 4,718,592

    k_prep<<<(COUT * KDEF + 255) / 256, 256, 0, stream>>>(pw_w, de_w, off_w,
                                                          pwA, wA, offA, stats);
    k_dw<<<(nElem + 255) / 256, 256, 0, stream>>>(x_in, dw_w, dw_b, h0);
    k_pw_wmma<<<dim3(PP / NT_PW, BATCH), 256, 0, stream>>>(h0, pwA, pw_b, sc);
    k_inorm<<<dim3(COUT, BATCH), 256, 0, stream>>>(sc, h_cl);
    k_offc_wmma<<<dim3(PP / 16, BATCH), 256, 0, stream>>>(h_cl, offA, off_b, offb);
    k_deform_wmma<<<dim3(PP / 16, BATCH), 256, 0, stream>>>(h_cl, offb, wA, de_b, dbuf);
    k_gstat<<<dim3(288, BATCH), 256, 0, stream>>>(dbuf, stats);
    k_gfin<<<1, 32, 0, stream>>>(stats, mr);
    k_gate<<<(nElem + 255) / 256, 256, 0, stream>>>(dbuf, sc, gn_w, gn_b, mr, out);
}
